// SpatialSelfAttention_4294967296477
// MI455X (gfx1250) — matrix operand order repaired and checked by reading, not compiled, not run on hardware
//
#include <hip/hip_runtime.h>
#include <hip/hip_bf16.h>

// ---------------------------------------------------------------------------
// Shifted-window attention for MI455X (gfx1250), wave32, v_wmma_f32_16x16x32_f16.
//   B=16, C=256, H=W=64, heads=8, head_dim=32, window=8x8 (64 tok), shift=4.
// All fragment traffic is 16B LDS accesses (ds_load_b128):
//   A frags: row-major [m][k], two contiguous 8-half runs per lane.
//   B frags: staged transposed [n][k], two contiguous 8-half runs per lane.
// Attention computes S^T = K*Q^T and Out^T = V^T*P^T so D-fragments are
// per-lane contiguous and store as single b128s.
// GEMM kernels use wide M-tiles (384 / 256) so activations are read with
// minimal redundancy and B-fragments are reused across M-tiles in registers.
// ---------------------------------------------------------------------------

typedef __attribute__((ext_vector_type(16))) _Float16 v16h;
typedef __attribute__((ext_vector_type(8)))  _Float16 v8h;
typedef __attribute__((ext_vector_type(4)))  _Float16 v4h;
typedef __attribute__((ext_vector_type(8)))  float    v8f;

__device__ __forceinline__ v8f wmma_f16(v16h a, v16h b, v8f c) {
  return __builtin_amdgcn_wmma_f32_16x16x32_f16(false, a, false, b, (short)0, c,
                                                false, false);
}

// A fragment, A row-major [m][k] (strideM mult of 8 halves, base 16B aligned).
// Lane L: m=L&15; k runs {0..7}+8*half and {16..23}+8*half  -> 2x ds_load_b128.
__device__ __forceinline__ v16h frag_a_rm(const _Float16* base, int strideM, int lane) {
  const _Float16* p = base + (lane & 15) * strideM + ((lane >> 4) << 3);
  v8h lo = *(const v8h*)p;
  v8h hi = *(const v8h*)(p + 16);
  return __builtin_shufflevector(lo, hi, 0, 1, 2, 3, 4, 5, 6, 7,
                                 8, 9, 10, 11, 12, 13, 14, 15);
}

// B fragment, B staged k-major per column: element (k,n) = base[n*strideN + k].
// Lane L: n=L&15; k runs {0..7}+8*half and {16..23}+8*half -> 2x ds_load_b128.
__device__ __forceinline__ v16h frag_b_kmaj(const _Float16* base, int strideN, int lane) {
  const _Float16* p = base + (lane & 15) * strideN + ((lane >> 4) << 3);
  v8h lo = *(const v8h*)p;
  v8h hi = *(const v8h*)(p + 16);
  return __builtin_shufflevector(lo, hi, 0, 1, 2, 3, 4, 5, 6, 7,
                                 8, 9, 10, 11, 12, 13, 14, 15);
}

#define APAD 40   // [m][k] chunk stride (halves): mult of 8, 20 dwords -> clean banks

// ---------------------------------------------------------------------------
// Kernel 1: QKV = W_qkv[768,256] * X_shifted[256,4096] per batch.
// Grid (2 m-tiles of 384, 64 shifted rows, 16 batches), 256 thr = 8 waves.
// Per wave: 3 M-subtiles x 4 N-tiles, B-frags cached in regs across M.
// Q,K -> qkws[win][2][tok][c] (b128 stores); V -> vtws[win][c][tok] (transposed).
// ---------------------------------------------------------------------------
__global__ __launch_bounds__(256) void qkv_kernel(const float* __restrict__ x,
                                                  const float* __restrict__ w_qkv,
                                                  const int* __restrict__ shift,
                                                  _Float16* __restrict__ qkws,
                                                  _Float16* __restrict__ vtws) {
  __shared__ __align__(16) _Float16 As[384 * APAD];   // W chunk, [m][k]
  __shared__ __align__(16) _Float16 Bst[64 * APAD];   // X chunk, transposed [n][k]
  const int tid = threadIdx.x, lane = tid & 31, wave = tid >> 5;
  const int bm = blockIdx.x;           // 0..1 -> o base = bm*384
  const int h  = blockIdx.y;           // shifted-space row
  const int b  = blockIdx.z;
  const int s  = (*shift != 0) ? 4 : 0;
  const int hsrc = (h + s) & 63;

  v8f acc[3][4] = {};
  for (int k0 = 0; k0 < 256; k0 += 32) {
    for (int i = tid; i < 384 * 8; i += 256) {           // A: v4h packed stores
      int mm = i >> 3, k4 = i & 7;
      const float* src = &w_qkv[(mm + bm * 384) * 256 + k0 + k4 * 4];
      v4h pk;
#pragma unroll
      for (int j = 0; j < 4; ++j) pk[j] = (_Float16)src[j];
      *(v4h*)&As[mm * APAD + k4 * 4] = pk;
    }
    for (int i = tid; i < 64 * 8; i += 256) {            // B^T: v4h packed stores
      int k4 = i >> 6, n = i & 63;
      int wsrc = (n + s) & 63;                           // roll(-s) folded in
      const float* src = &x[(size_t)(b * 256 + k0 + k4 * 4) * 4096 + hsrc * 64 + wsrc];
      v4h pk;
#pragma unroll
      for (int j = 0; j < 4; ++j) pk[j] = (_Float16)src[(size_t)j * 4096];
      *(v4h*)&Bst[n * APAD + k4 * 4] = pk;
    }
    __syncthreads();
    v16h bf[4];
#pragma unroll
    for (int ni = 0; ni < 4; ++ni) bf[ni] = frag_b_kmaj(Bst + ni * 16 * APAD, APAD, lane);
#pragma unroll
    for (int mi = 0; mi < 3; ++mi) {
      v16h af = frag_a_rm(As + (wave * 3 + mi) * 16 * APAD, APAD, lane);
#pragma unroll
      for (int ni = 0; ni < 4; ++ni) acc[mi][ni] = wmma_f16(af, bf[ni], acc[mi][ni]);
    }
    __syncthreads();
  }

  const int halfid = lane >> 4, nrel = lane & 15;
#pragma unroll
  for (int mi = 0; mi < 3; ++mi) {
    int o0  = bm * 384 + (wave * 3 + mi) * 16 + halfid * 8;
    int mat = o0 >> 8;                                   // uniform per tile
    int c0  = o0 & 255;
#pragma unroll
    for (int ni = 0; ni < 4; ++ni) {
      int wcol = ni * 16 + nrel;
      int win  = b * 64 + (h >> 3) * 8 + (wcol >> 3);
      int tok  = (h & 7) * 8 + (wcol & 7);
      if (mat < 2) {                                     // Q/K: one b128 store
        v8h pk;
#pragma unroll
        for (int r = 0; r < 8; ++r) pk[r] = (_Float16)acc[mi][ni][r];
        *(v8h*)&qkws[(((size_t)win * 2 + mat) * 64 + tok) * 256 + c0] = pk;
      } else {                                           // V: transposed scatter
#pragma unroll
        for (int r = 0; r < 8; ++r)
          vtws[(size_t)win * 16384 + (c0 + r) * 64 + tok] = (_Float16)acc[mi][ni][r];
      }
    }
  }
}

// ---------------------------------------------------------------------------
// Kernel 2: per-window MHA. 1 block/window, 1 wave/head.
//   S^T = K*Q^T (scaled, b128-packed into P[t][t']) -> softmax -> Out^T = V^T*P^T
// ---------------------------------------------------------------------------
#define QSTR 264   // Q/K row stride (halves): mult of 8, 132 dwords
#define VSTR 72    // Vt row stride: mult of 8, 36 dwords
#define PSTR 72    // P  row stride

__global__ __launch_bounds__(256) void attn_kernel(const _Float16* __restrict__ qkws,
                                                   const _Float16* __restrict__ vtws,
                                                   _Float16* __restrict__ yws) {
  extern __shared__ _Float16 smem[];
  _Float16* qk = smem;                   // [2][64][QSTR]
  _Float16* Vt = smem + 2 * 64 * QSTR;   // [256][VSTR]   (V transposed [c][t])
  _Float16* P  = Vt + 256 * VSTR;        // [8][64][PSTR] (probs, row-major [t][t'])
  const int tid = threadIdx.x, lane = tid & 31, nh = tid >> 5;
  const int win = blockIdx.x;

  {  // stage Q,K (uint4) and Vt (uint2)
    const uint4* qs = (const uint4*)(qkws + (size_t)win * 2 * 64 * 256);
    for (int i = tid; i < 2 * 64 * 32; i += 256) {
      int row = i >> 5, q4 = i & 31;
      *(uint4*)((uint*)qk + row * (QSTR / 2) + q4 * 4) = qs[i];
    }
    const uint2* vs = (const uint2*)(vtws + (size_t)win * 16384);
    for (int i = tid; i < 256 * 16; i += 256) {
      int c = i >> 4, d2 = i & 15;
      *(uint2*)((uint*)Vt + c * (VSTR / 2) + d2 * 2) = vs[i];
    }
  }
  __syncthreads();

  const _Float16* qb = qk + nh * 32;               // Q [t][c] head slice
  const _Float16* kb = qk + 64 * QSTR + nh * 32;   // K [t][c] head slice
  _Float16* Ph = P + nh * 64 * PSTR;
  const int halfid = lane >> 4, nrel = lane & 15;
  const float scale = 0.17677669529663687f;        // 1/sqrt(32)

  // ---- S^T = K * Q^T (A = K rows t', B = Q^T k-major); D packs to P[t][t'] ----
#pragma unroll
  for (int mi = 0; mi < 4; ++mi) {                 // t' tiles
    v16h af = frag_a_rm(kb + mi * 16 * QSTR, QSTR, lane);
#pragma unroll
    for (int ni = 0; ni < 4; ++ni) {               // t tiles
      v16h bf = frag_b_kmaj(qb + ni * 16 * QSTR, QSTR, lane);
      v8f acc = {};
      acc = wmma_f16(af, bf, acc);
      v8h pk;
#pragma unroll
      for (int r = 0; r < 8; ++r) pk[r] = (_Float16)(acc[r] * scale);
      *(v8h*)&Ph[(ni * 16 + nrel) * PSTR + mi * 16 + halfid * 8] = pk;
    }
  }
  __syncthreads();

  // ---- softmax over each row of P (lane owns 2 rows; v8h chunked) ----
  for (int row = lane; row < 64; row += 32) {
    _Float16* pr = Ph + row * PSTR;
    v8h ch[8];
    float mx = -1e30f;
#pragma unroll
    for (int jj = 0; jj < 8; ++jj) {
      ch[jj] = *(const v8h*)(pr + jj * 8);
#pragma unroll
      for (int e = 0; e < 8; ++e) mx = fmaxf(mx, (float)ch[jj][e]);
    }
    float sum = 0.f;
#pragma unroll
    for (int jj = 0; jj < 8; ++jj)
#pragma unroll
      for (int e = 0; e < 8; ++e) {
        float ev = __expf((float)ch[jj][e] - mx);
        sum += ev;
        ch[jj][e] = (_Float16)ev;
      }
    float inv = 1.0f / sum;
#pragma unroll
    for (int jj = 0; jj < 8; ++jj) {
#pragma unroll
      for (int e = 0; e < 8; ++e) ch[jj][e] = (_Float16)((float)ch[jj][e] * inv);
      *(v8h*)(pr + jj * 8) = ch[jj];
    }
  }
  __syncthreads();

  // ---- Out^T = V^T * P^T (A = Vt rows c, B = P[t][t'] k-major over t') ----
#pragma unroll
  for (int mi = 0; mi < 2; ++mi) {                 // c tiles (32 per head)
    v16h af0 = frag_a_rm(Vt + (nh * 32 + mi * 16) * VSTR + 0,  VSTR, lane);
    v16h af1 = frag_a_rm(Vt + (nh * 32 + mi * 16) * VSTR + 32, VSTR, lane);
#pragma unroll
    for (int ni = 0; ni < 4; ++ni) {               // t tiles
      v8f acc = {};
      v16h bf0 = frag_b_kmaj(Ph + ni * 16 * PSTR + 0,  PSTR, lane);
      acc = wmma_f16(af0, bf0, acc);
      v16h bf1 = frag_b_kmaj(Ph + ni * 16 * PSTR + 32, PSTR, lane);
      acc = wmma_f16(af1, bf1, acc);
      int t  = ni * 16 + nrel;
      int c0 = nh * 32 + mi * 16 + halfid * 8;
      v8h pk;
#pragma unroll
      for (int r = 0; r < 8; ++r) pk[r] = (_Float16)acc[r];
      *(v8h*)&yws[((size_t)win * 64 + t) * 256 + c0] = pk;   // one b128 store
    }
  }
}

// ---------------------------------------------------------------------------
// Kernel 3: out = W_proj[256,256] * window_reverse(unroll(y)). fp32 d_out.
// Grid (64 rows, 16 batches); full M=256 per block, 2 M-subtiles per wave.
// ---------------------------------------------------------------------------
__global__ __launch_bounds__(256) void proj_kernel(const _Float16* __restrict__ yws,
                                                   const float* __restrict__ w_proj,
                                                   const int* __restrict__ shift,
                                                   float* __restrict__ out) {
  __shared__ __align__(16) _Float16 As[256 * APAD];
  __shared__ __align__(16) _Float16 Bst[64 * APAD];
  const int tid = threadIdx.x, lane = tid & 31, wave = tid >> 5;
  const int h = blockIdx.x;            // final-space row
  const int b = blockIdx.y;
  const int s = (*shift != 0) ? 4 : 0;
  const int hh = (h - s + 64) & 63;    // shifted-space row

  v8f acc[2][4] = {};
  for (int k0 = 0; k0 < 256; k0 += 32) {
    for (int i = tid; i < 256 * 8; i += 256) {
      int mm = i >> 3, k4 = i & 7;
      const float* src = &w_proj[mm * 256 + k0 + k4 * 4];
      v4h pk;
#pragma unroll
      for (int j = 0; j < 4; ++j) pk[j] = (_Float16)src[j];
      *(v4h*)&As[mm * APAD + k4 * 4] = pk;
    }
    for (int i = tid; i < 64 * 8; i += 256) {      // y is f16: b64 load + b64 store
      int n = i >> 3, k4 = i & 7;
      int ww  = (n - s + 64) & 63;                 // roll(+s) folded in
      int win = b * 64 + (hh >> 3) * 8 + (ww >> 3);
      int tok = (hh & 7) * 8 + (ww & 7);
      *(v4h*)&Bst[n * APAD + k4 * 4] =
          *(const v4h*)&yws[((size_t)win * 64 + tok) * 256 + k0 + k4 * 4];
    }
    __syncthreads();
    v16h bf[4];
#pragma unroll
    for (int ni = 0; ni < 4; ++ni) bf[ni] = frag_b_kmaj(Bst + ni * 16 * APAD, APAD, lane);
#pragma unroll
    for (int mi = 0; mi < 2; ++mi) {
      v16h af = frag_a_rm(As + (wave * 2 + mi) * 16 * APAD, APAD, lane);
#pragma unroll
      for (int ni = 0; ni < 4; ++ni) acc[mi][ni] = wmma_f16(af, bf[ni], acc[mi][ni]);
    }
    __syncthreads();
  }

  const int halfid = lane >> 4, nrel = lane & 15;
#pragma unroll
  for (int mi = 0; mi < 2; ++mi)
#pragma unroll
    for (int ni = 0; ni < 4; ++ni) {
      int w = ni * 16 + nrel;
#pragma unroll
      for (int r = 0; r < 8; ++r) {
        int o = (wave * 2 + mi) * 16 + halfid * 8 + r;
        out[(((size_t)b * 256 + o) * 64 + h) * 64 + w] = acc[mi][ni][r];
      }
    }
}

// ---------------------------------------------------------------------------
extern "C" void kernel_launch(void* const* d_in, const int* in_sizes, int n_in,
                              void* d_out, int out_size, void* d_ws, size_t ws_size,
                              hipStream_t stream) {
  const float* x      = (const float*)d_in[0];   // [16,256,64,64]
  const float* w_qkv  = (const float*)d_in[1];   // [768,256]
  const float* w_proj = (const float*)d_in[2];   // [256,256]
  const int*   shift  = (const int*)d_in[3];     // scalar

  _Float16* qkws = (_Float16*)d_ws;                       // [1024][2][64][256] (Q,K)
  _Float16* vtws = qkws + (size_t)1024 * 2 * 64 * 256;    // [1024][256][64]    (V^T)
  _Float16* yws  = vtws + (size_t)1024 * 256 * 64;        // [1024][64][256]    (attn out)
  float*    out  = (float*)d_out;

  qkv_kernel<<<dim3(2, 64, 16), 256, 0, stream>>>(x, w_qkv, shift, qkws, vtws);

  size_t lds2 = (size_t)(2 * 64 * QSTR + 256 * VSTR + 8 * 64 * PSTR) * sizeof(_Float16);
  attn_kernel<<<dim3(1024), 256, lds2, stream>>>(qkws, vtws, yws);

  proj_kernel<<<dim3(64, 16), 256, 0, stream>>>(yws, w_proj, shift, out);
}
